// MultiHeadSelfAttention_70738111365311
// MI455X (gfx1250) — matrix operand order repaired and checked by reading, not compiled, not run on hardware
//
#include <hip/hip_runtime.h>
#include <hip/hip_bf16.h>

typedef __bf16 bf16_t;
typedef __attribute__((ext_vector_type(16))) __bf16 v16bf;
typedef __attribute__((ext_vector_type(8)))  float  v8f;

#define DIMX 384
#define D3   1152
#define NHX  6
#define HDX  64
#define BBX  16
#define NNX  1024
#define MTOT (BBX * NNX)

// ---- fragment load helpers ------------------------------------------------
// B-frag (32x16, lane = column): K split {0-7,16-23} / {8-15,24-31}; chunks at +0 and +16
__device__ __forceinline__ v16bf load_b_frag(const bf16_t* p) {
    v16bf b;
    *(uint4*)&b       = *(const uint4*)(p);
    *((uint4*)&b + 1) = *(const uint4*)(p + 16);
    return b;
}
// A-frag (16x32, lane = row): K split {0-7,16-23} / {8-15,24-31}; p already
// includes the per-half sub-offset, so chunks are at +0 and +16.
__device__ __forceinline__ v16bf load_a_frag_bf(const bf16_t* p) {
    v16bf a;
    *(uint4*)&a       = *(const uint4*)(p);
    *((uint4*)&a + 1) = *(const uint4*)(p + 16);
    return a;
}
__device__ __forceinline__ v16bf load_a_frag_f32(const float* p) {
    float4 f0 = *(const float4*)(p);
    float4 f1 = *(const float4*)(p + 4);
    float4 g0 = *(const float4*)(p + 16);
    float4 g1 = *(const float4*)(p + 20);
    v16bf a;
    a[0]=(bf16_t)f0.x;  a[1]=(bf16_t)f0.y;  a[2]=(bf16_t)f0.z;  a[3]=(bf16_t)f0.w;
    a[4]=(bf16_t)f1.x;  a[5]=(bf16_t)f1.y;  a[6]=(bf16_t)f1.z;  a[7]=(bf16_t)f1.w;
    a[8]=(bf16_t)g0.x;  a[9]=(bf16_t)g0.y;  a[10]=(bf16_t)g0.z; a[11]=(bf16_t)g0.w;
    a[12]=(bf16_t)g1.x; a[13]=(bf16_t)g1.y; a[14]=(bf16_t)g1.z; a[15]=(bf16_t)g1.w;
    return a;
}
#define WMMA_BF16(A, B, C) \
    __builtin_amdgcn_wmma_f32_16x16x32_bf16(false, (A), false, (B), (short)0, (C), false, false)

// ---------------------------------------------------------------------------
// Kernel 1: convert + transpose weights to bf16 column-major (B-frag friendly)
// ---------------------------------------------------------------------------
__global__ __launch_bounds__(256) void prep_weights(const float* __restrict__ wqkv,
                                                    const float* __restrict__ wproj,
                                                    bf16_t* __restrict__ wqkvT,
                                                    bf16_t* __restrict__ wprojT) {
    int idx = blockIdx.x * 256 + threadIdx.x;
    const int t1 = DIMX * D3;
    const int t2 = DIMX * DIMX;
    if (idx < t1) {
        int n = idx / DIMX, k = idx - n * DIMX;
        wqkvT[idx] = (bf16_t)wqkv[k * D3 + n];          // wT[n][k] = w[k][n]
    } else if (idx < t1 + t2) {
        int j = idx - t1;
        int n = j / DIMX, k = j - n * DIMX;
        wprojT[j] = (bf16_t)wproj[k * DIMX + n];
    }
}

// ---------------------------------------------------------------------------
// Kernel 2: qkv = x @ w_qkv + b_qkv  (bf16 WMMA, f32 accum)
// Ping-pong double buffer over K (no copy-backs), 6 iterations of 2 phases.
// Writes q (pre-scaled), k as [b][h][n][d], v transposed [b][h][d][n]
// ---------------------------------------------------------------------------
__global__ __launch_bounds__(256) void qkv_gemm(const float* __restrict__ x,
                                                const bf16_t* __restrict__ wT,
                                                const float* __restrict__ bias,
                                                bf16_t* __restrict__ qb,
                                                bf16_t* __restrict__ kb,
                                                bf16_t* __restrict__ vtb) {
    const int lane  = threadIdx.x & 31;
    const int wave  = threadIdx.x >> 5;
    const int ln16  = lane & 15;
    const int hi    = lane >> 4;
    const int akoff = hi * 8;
    const int bkoff = hi * 8;

    const int m0 = blockIdx.x * 128 + wave * 16;
    const int n0 = blockIdx.y * 64;

    const float*  __restrict__ arow  = x  + (size_t)(m0 + ln16) * DIMX + akoff;
    const bf16_t* __restrict__ wbase = wT + (size_t)(n0 + ln16) * DIMX + bkoff;

    v8f acc[4] = {};

    v16bf a0, a1, b0[4], b1[4];
    a0 = load_a_frag_f32(arow);
#pragma unroll
    for (int t = 0; t < 4; ++t) b0[t] = load_b_frag(wbase + (size_t)t * 16 * DIMX);

    for (int k0 = 0; k0 < DIMX; k0 += 64) {
        // phase 0: load k0+32, consume k0
        a1 = load_a_frag_f32(arow + k0 + 32);
#pragma unroll
        for (int t = 0; t < 4; ++t) b1[t] = load_b_frag(wbase + (size_t)t * 16 * DIMX + k0 + 32);
#pragma unroll
        for (int t = 0; t < 4; ++t) acc[t] = WMMA_BF16(a0, b0[t], acc[t]);
        // phase 1: load k0+64 (clamped; harmless reload on last), consume k0+32
        const int kn = (k0 + 64 < DIMX) ? (k0 + 64) : 0;
        a0 = load_a_frag_f32(arow + kn);
#pragma unroll
        for (int t = 0; t < 4; ++t) b0[t] = load_b_frag(wbase + (size_t)t * 16 * DIMX + kn);
#pragma unroll
        for (int t = 0; t < 4; ++t) acc[t] = WMMA_BF16(a1, b1[t], acc[t]);
    }

    // Epilogue: bias, split into q/k/vT, convert to bf16
#pragma unroll
    for (int t = 0; t < 4; ++t) {
        const int col = n0 + t * 16 + ln16;
        const float bv = bias[col];
#pragma unroll
        for (int i = 0; i < 8; ++i) {
            const int r    = i + hi * 8;
            const int mtok = m0 + r;
            const int bi   = mtok >> 10;
            const int ni   = mtok & (NNX - 1);
            float val = acc[t][i] + bv;
            if (col < DIMX) {                              // Q (fold softmax scale)
                int h = col / HDX, d = col % HDX;
                qb[(((size_t)bi * NHX + h) * NNX + ni) * HDX + d] = (bf16_t)(val * 0.125f);
            } else if (col < 2 * DIMX) {                   // K
                int cc = col - DIMX; int h = cc / HDX, d = cc % HDX;
                kb[(((size_t)bi * NHX + h) * NNX + ni) * HDX + d] = (bf16_t)val;
            } else {                                       // V transposed [b][h][d][n]
                int cc = col - 2 * DIMX; int h = cc / HDX, d = cc % HDX;
                vtb[(((size_t)bi * NHX + h) * HDX + d) * NNX + ni] = (bf16_t)val;
            }
        }
    }
}

// ---------------------------------------------------------------------------
// Kernel 3: flash attention per (b,h). Block = 128 q-rows (8 waves x 16),
// streams keys in chunks of 64. K-frags batch-preloaded; V-frags issued
// before softmax so they overlap the VALU-heavy section.
// ---------------------------------------------------------------------------
__global__ __launch_bounds__(256) void flash_attn(const bf16_t* __restrict__ qb,
                                                  const bf16_t* __restrict__ kb,
                                                  const bf16_t* __restrict__ vtb,
                                                  bf16_t* __restrict__ ob) {
    __shared__ float plds[8][16][68];   // per-wave P staging (C-layout -> A-layout)

    const int lane  = threadIdx.x & 31;
    const int wave  = threadIdx.x >> 5;
    const int ln16  = lane & 15;
    const int hi    = lane >> 4;
    const int akoff = hi * 8;
    const int bkoff = hi * 8;

    const int head = blockIdx.y;        // 0..95
    const int bi   = head / NHX;
    const int hh   = head % NHX;
    const int q0   = blockIdx.x * 128 + wave * 16;

    const bf16_t* __restrict__ kbase = kb  + (size_t)head * NNX * HDX;
    const bf16_t* __restrict__ vbase = vtb + (size_t)head * HDX * NNX;

    // Q A-fragments for both K-steps (hd = 64 -> 2 steps of 32), kept in regs
    v16bf qf[2];
    {
        const bf16_t* qrow = qb + ((size_t)head * NNX + (q0 + ln16)) * HDX + akoff;
#pragma unroll
        for (int s = 0; s < 2; ++s) qf[s] = load_a_frag_bf(qrow + s * 32);
    }

    v8f o[4] = {};
    float mrow[8], lrow[8];
#pragma unroll
    for (int i = 0; i < 8; ++i) { mrow[i] = -3.0e38f; lrow[i] = 0.0f; }

    for (int c = 0; c < NNX; c += 64) {
        if (c + 64 < NNX)   // prefetch next K chunk (global_prefetch_b8)
            __builtin_prefetch(kbase + (size_t)(c + 64) * HDX + lane * 128, 0, 1);

        // ---- batch-load all K fragments for this chunk (one big clause) ----
        v16bf kf[4][2];
#pragma unroll
        for (int t = 0; t < 4; ++t) {
            const bf16_t* krow = kbase + (size_t)(c + t * 16 + ln16) * HDX + bkoff;
#pragma unroll
            for (int s = 0; s < 2; ++s) kf[t][s] = load_b_frag(krow + s * 32);
        }

        // ---- S = Q @ K^T (scale pre-folded into Q) ----
        v8f s4[4] = {};
#pragma unroll
        for (int t = 0; t < 4; ++t)
#pragma unroll
            for (int s = 0; s < 2; ++s)
                s4[t] = WMMA_BF16(qf[s], kf[t][s], s4[t]);

        // ---- issue V fragment loads now; they overlap the softmax VALU ----
        v16bf vf[2][4];
#pragma unroll
        for (int s = 0; s < 2; ++s)
#pragma unroll
            for (int t = 0; t < 4; ++t) {
                const bf16_t* vrow = vbase + (size_t)(t * 16 + ln16) * NNX + c + s * 32 + bkoff;
                vf[s][t] = load_b_frag(vrow);
            }

        // ---- online softmax (row = acc slot, cols = 16 lanes of half-wave) ----
        float mnew[8];
#pragma unroll
        for (int i = 0; i < 8; ++i) {
            float v = fmaxf(fmaxf(s4[0][i], s4[1][i]), fmaxf(s4[2][i], s4[3][i]));
            v = fmaxf(v, __shfl_xor(v, 1, 32));
            v = fmaxf(v, __shfl_xor(v, 2, 32));
            v = fmaxf(v, __shfl_xor(v, 4, 32));
            v = fmaxf(v, __shfl_xor(v, 8, 32));
            mnew[i] = fmaxf(mrow[i], v);
        }
#pragma unroll
        for (int t = 0; t < 4; ++t)
#pragma unroll
            for (int i = 0; i < 8; ++i)
                s4[t][i] = __expf(s4[t][i] - mnew[i]);
#pragma unroll
        for (int i = 0; i < 8; ++i) {
            float v = s4[0][i] + s4[1][i] + s4[2][i] + s4[3][i];
            v += __shfl_xor(v, 1, 32);
            v += __shfl_xor(v, 2, 32);
            v += __shfl_xor(v, 4, 32);
            v += __shfl_xor(v, 8, 32);
            float alpha = __expf(mrow[i] - mnew[i]);
            lrow[i] = lrow[i] * alpha + v;
            mrow[i] = mnew[i];
#pragma unroll
            for (int t = 0; t < 4; ++t) o[t][i] *= alpha;
        }

        // ---- stage P to per-wave LDS (lane-transpose C-layout -> A-layout) ----
#pragma unroll
        for (int t = 0; t < 4; ++t)
#pragma unroll
            for (int i = 0; i < 8; ++i)
                plds[wave][i + hi * 8][t * 16 + ln16] = s4[t][i];
        // LDS ops are in-order within a wave; region is wave-private -> no barrier

        // ---- O += P @ V (V fragments already resident) ----
#pragma unroll
        for (int s = 0; s < 2; ++s) {
            v16bf pf;
            const float* pr = &plds[wave][ln16][s * 32 + akoff];
#pragma unroll
            for (int e = 0; e < 8; ++e) pf[e]     = (bf16_t)pr[e];
#pragma unroll
            for (int e = 0; e < 8; ++e) pf[8 + e] = (bf16_t)pr[16 + e];
#pragma unroll
            for (int t = 0; t < 4; ++t)
                o[t] = WMMA_BF16(pf, vf[s][t], o[t]);
        }
    }

    // ---- normalize and write attention output as (B, N, D) bf16 ----
    float inv[8];
#pragma unroll
    for (int i = 0; i < 8; ++i) inv[i] = 1.0f / lrow[i];
#pragma unroll
    for (int t = 0; t < 4; ++t) {
        const int col = hh * HDX + t * 16 + ln16;
#pragma unroll
        for (int i = 0; i < 8; ++i) {
            const int r = i + hi * 8;
            ob[((size_t)bi * NNX + (q0 + r)) * DIMX + col] = (bf16_t)(o[t][i] * inv[i]);
        }
    }
}

// ---------------------------------------------------------------------------
// Kernel 4: out = attn @ w_proj + b_proj  (bf16 WMMA, ping-pong K buffer)
// ---------------------------------------------------------------------------
__global__ __launch_bounds__(256) void proj_gemm(const bf16_t* __restrict__ ab,
                                                 const bf16_t* __restrict__ wT,
                                                 const float* __restrict__ bias,
                                                 float* __restrict__ out) {
    const int lane  = threadIdx.x & 31;
    const int wave  = threadIdx.x >> 5;
    const int ln16  = lane & 15;
    const int hi    = lane >> 4;
    const int akoff = hi * 8;
    const int bkoff = hi * 8;
    const int m0 = blockIdx.x * 128 + wave * 16;
    const int n0 = blockIdx.y * 64;

    const bf16_t* __restrict__ arow  = ab + (size_t)(m0 + ln16) * DIMX + akoff;
    const bf16_t* __restrict__ wbase = wT + (size_t)(n0 + ln16) * DIMX + bkoff;

    v8f acc[4] = {};

    v16bf a0, a1, b0[4], b1[4];
    a0 = load_a_frag_bf(arow);
#pragma unroll
    for (int t = 0; t < 4; ++t) b0[t] = load_b_frag(wbase + (size_t)t * 16 * DIMX);

    for (int k0 = 0; k0 < DIMX; k0 += 64) {
        a1 = load_a_frag_bf(arow + k0 + 32);
#pragma unroll
        for (int t = 0; t < 4; ++t) b1[t] = load_b_frag(wbase + (size_t)t * 16 * DIMX + k0 + 32);
#pragma unroll
        for (int t = 0; t < 4; ++t) acc[t] = WMMA_BF16(a0, b0[t], acc[t]);
        const int kn = (k0 + 64 < DIMX) ? (k0 + 64) : 0;
        a0 = load_a_frag_bf(arow + kn);
#pragma unroll
        for (int t = 0; t < 4; ++t) b0[t] = load_b_frag(wbase + (size_t)t * 16 * DIMX + kn);
#pragma unroll
        for (int t = 0; t < 4; ++t) acc[t] = WMMA_BF16(a1, b1[t], acc[t]);
    }

#pragma unroll
    for (int t = 0; t < 4; ++t) {
        const int col = n0 + t * 16 + ln16;
        const float bv = bias[col];
#pragma unroll
        for (int i = 0; i < 8; ++i) {
            const int r = i + hi * 8;
            out[(size_t)(m0 + r) * DIMX + col] = acc[t][i] + bv;
        }
    }
}

// ---------------------------------------------------------------------------
extern "C" void kernel_launch(void* const* d_in, const int* in_sizes, int n_in,
                              void* d_out, int out_size, void* d_ws, size_t ws_size,
                              hipStream_t stream) {
    const float* x     = (const float*)d_in[0];
    const float* wqkv  = (const float*)d_in[1];
    const float* bqkv  = (const float*)d_in[2];
    const float* wproj = (const float*)d_in[3];
    const float* bproj = (const float*)d_in[4];
    float* out = (float*)d_out;

    char* ws = (char*)d_ws;
    const size_t sz_head = (size_t)BBX * NHX * NNX * HDX * sizeof(bf16_t); // 12.58 MB
    bf16_t* qb     = (bf16_t*)(ws);
    bf16_t* kbuf   = (bf16_t*)(ws + sz_head);
    bf16_t* vtb    = (bf16_t*)(ws + 2 * sz_head);
    bf16_t* attn   = (bf16_t*)(ws + 3 * sz_head);
    bf16_t* wqkvT  = (bf16_t*)(ws + 4 * sz_head);
    bf16_t* wprojT = (bf16_t*)(ws + 4 * sz_head + (size_t)D3 * DIMX * sizeof(bf16_t));

    {
        int total = DIMX * D3 + DIMX * DIMX;
        prep_weights<<<(total + 255) / 256, 256, 0, stream>>>(wqkv, wproj, wqkvT, wprojT);
    }
    {
        dim3 grid(MTOT / 128, D3 / 64);
        qkv_gemm<<<grid, 256, 0, stream>>>(x, wqkvT, bqkv, qb, kbuf, vtb);
    }
    {
        dim3 grid(NNX / 128, BBX * NHX);
        flash_attn<<<grid, 256, 0, stream>>>(qb, kbuf, vtb, attn);
    }
    {
        dim3 grid(MTOT / 128, DIMX / 64);
        proj_gemm<<<grid, 256, 0, stream>>>(attn, wprojT, bproj, out);
    }
}
